// SimpleMultiHeadAttention_75703093559313
// MI455X (gfx1250) — matrix operand order repaired and checked by reading, not compiled, not run on hardware
//
#include <hip/hip_runtime.h>
#include <hip/hip_bf16.h>
#include <stdint.h>

// Problem constants
#define SEQ   2048
#define NB    2          // batch
#define DM    1024
#define NH    16
#define DHD   64
#define MROWS (SEQ*NB)   // 4096

typedef __attribute__((ext_vector_type(16))) _Float16 v16h;
typedef __attribute__((ext_vector_type(8)))  float    v8f;
typedef int v4i __attribute__((vector_size(16)));   // matches builtin param type
typedef _Float16 f16;

#define GAS __attribute__((address_space(1)))
#define LAS __attribute__((address_space(3)))

union Frag16 {
  v16h v;
  uint32_t u[8];
  uint4    q[2];
};

__device__ __forceinline__ v8f wmma_f16f32(const Frag16& a, const Frag16& b, v8f c) {
  return __builtin_amdgcn_wmma_f32_16x16x32_f16(false, a.v, false, b.v,
                                                (short)0, c, false, false);
}

// Async global->LDS 16B copy (ASYNCcnt-tracked, no VGPR round trip).
// Low 32 bits of a generic pointer to LDS are the LDS byte offset.
__device__ __forceinline__ void async_b128(const void* g, const f16* l) {
  __builtin_amdgcn_global_load_async_to_lds_b128(
      (GAS v4i*)(uintptr_t)g,
      (LAS v4i*)(uint32_t)(uintptr_t)l, 0, 0);
}
__device__ __forceinline__ void wait_async0() {
  __builtin_amdgcn_s_wait_asynccnt(0);
}

// A fragment (16x32 f16, ISA 7.12.2 layout): lane L holds row m=L&15,
// hi lanes hold K+8; VGPR p holds K pair base (p<4 ? 2p : 16+2(p-4)).
// LDS A tile is row-major with row stride 32 halves.
__device__ __forceinline__ void load_a_frag(const f16* As, int mrow, int hi, Frag16& a) {
  const f16* base = As + mrow * 32;
#pragma unroll
  for (int p = 0; p < 8; ++p) {
    int k = ((p < 4) ? (2 * p) : (16 + 2 * (p - 4))) + 8 * hi;
    a.u[p] = *(const uint32_t*)(base + k);
  }
}

// B fragment (32x16 f16): lane holds column n=lane&15; element s holds k in A's order
// (s<8 ? s+8*hi : 16+8*hi+(s-8)).
template <int LDB>
__device__ __forceinline__ void load_b_frag(const f16* Bs, int lane, int nbase, Frag16& b) {
  const f16* base = Bs + nbase + (lane & 15);
  const int hi = lane >> 4;
#pragma unroll
  for (int s = 0; s < 16; ++s) {
    int k = ((s < 8) ? s : (16 + (s - 8))) + 8 * hi;
    b.v[s] = base[k * LDB];
  }
}

// ---------------------------------------------------------------------------
// fp32 -> f16 pack, 4 elements/thread
__global__ __launch_bounds__(256) void cvt_f32_f16x4_kernel(const float4* __restrict__ src,
                                                            uint2* __restrict__ dst) {
  int i = blockIdx.x * 256 + threadIdx.x;
  float4 f = src[i];
  union { f16 h[4]; uint2 u; } pk;
  pk.h[0] = (f16)f.x; pk.h[1] = (f16)f.y; pk.h[2] = (f16)f.z; pk.h[3] = (f16)f.w;
  dst[i] = pk.u;
}

// W[n][k] fp32 -> WT[k][n] f16 (so GEMM B tiles are straight k-major copies)
__global__ __launch_bounds__(256) void transpose_pack_kernel(const float* __restrict__ W,
                                                             f16* __restrict__ WT) {
  __shared__ float t[32][33];
  int tx = threadIdx.x & 31, ty = threadIdx.x >> 5;  // 32 x 8
#pragma unroll
  for (int j = 0; j < 4; ++j) {
    int nrow = blockIdx.y * 32 + ty + j * 8;
    int kcol = blockIdx.x * 32 + tx;
    t[ty + j * 8][tx] = W[(size_t)nrow * DM + kcol];
  }
  __syncthreads();
#pragma unroll
  for (int j = 0; j < 4; ++j) {
    int krow = blockIdx.x * 32 + ty + j * 8;
    int ncol = blockIdx.y * 32 + tx;
    WT[(size_t)krow * DM + ncol] = (f16)t[tx][ty + j * 8];
  }
}

// ---------------------------------------------------------------------------
// Projection GEMM: Y[m,n] = sum_k X[m,k]*WT[k,n] + bias[n]
// M=4096 N=1024 K=1024; block tile 128x128, BK=32; async double-buffered LDS.
// MODE 0: write f16 [H][B][S][Dh]   (Q, V)
// MODE 1: write f16 [H][B][Dh][S]   (K transposed for QK^T)
// MODE 2: write f32 [M][N] to d_out (final Wo projection)
template <int MODE>
__global__ __launch_bounds__(256) void gemm_proj_kernel(const f16* __restrict__ X,
                                                        const f16* __restrict__ WT,
                                                        const float* __restrict__ bias,
                                                        f16* __restrict__ outH,
                                                        float* __restrict__ outF) {
  __shared__ __align__(16) f16 As[2][128 * 32];
  __shared__ __align__(16) f16 Bs[2][32 * 128];
  const int tid = threadIdx.x;
  const int lane = tid & 31, wave = tid >> 5;
  const int waveM = wave >> 2, waveN = wave & 3;
  const int hi = lane >> 4;
  const int m0 = blockIdx.y * 128, n0 = blockIdx.x * 128;

  auto issueA = [&](int kt, int buf) {
#pragma unroll
    for (int l = 0; l < 2; ++l) {        // 512 x b128 = 128x32 halves
      int i = tid + l * 256;
      int r = i >> 2, c = (i & 3) * 8;
      async_b128(X + (size_t)(m0 + r) * DM + kt * 32 + c, &As[buf][r * 32 + c]);
    }
  };
  auto issueB = [&](int kt, int buf) {
#pragma unroll
    for (int l = 0; l < 2; ++l) {        // 512 x b128 = 32x128 halves
      int i = tid + l * 256;
      int r = i >> 4, c = (i & 15) * 8;
      async_b128(WT + (size_t)(kt * 32 + r) * DM + n0 + c, &Bs[buf][r * 128 + c]);
    }
  };

  const v8f vzero = {0, 0, 0, 0, 0, 0, 0, 0};
  v8f acc[4][2];
#pragma unroll
  for (int i = 0; i < 4; ++i)
#pragma unroll
    for (int j = 0; j < 2; ++j) acc[i][j] = vzero;

  issueA(0, 0);
  issueB(0, 0);
  const int NK = DM / 32;
  for (int kt = 0; kt < NK; ++kt) {
    wait_async0();
    __syncthreads();  // tile kt landed everywhere; everyone done with buf (kt+1)&1
    if (kt + 1 < NK) {
      issueA(kt + 1, (kt + 1) & 1);
      issueB(kt + 1, (kt + 1) & 1);
    }
    const f16* Ac = As[kt & 1];
    const f16* Bc = Bs[kt & 1];

    Frag16 a[4], bf[2];
#pragma unroll
    for (int mt = 0; mt < 4; ++mt)
      load_a_frag(Ac, waveM * 64 + mt * 16 + (lane & 15), hi, a[mt]);
#pragma unroll
    for (int nt = 0; nt < 2; ++nt)
      load_b_frag<128>(Bc, lane, waveN * 32 + nt * 16, bf[nt]);
#pragma unroll
    for (int mt = 0; mt < 4; ++mt)
#pragma unroll
      for (int nt = 0; nt < 2; ++nt)
        acc[mt][nt] = wmma_f16f32(a[mt], bf[nt], acc[mt][nt]);
  }

  // D layout: lanes 0-15 n=lane,m=r ; lanes 16-31 n=lane-16,m=8+r
#pragma unroll
  for (int mt = 0; mt < 4; ++mt)
#pragma unroll
    for (int nt = 0; nt < 2; ++nt) {
      int n = n0 + waveN * 32 + nt * 16 + (lane & 15);
      float bv = bias[n];
#pragma unroll
      for (int r = 0; r < 8; ++r) {
        int m = m0 + waveM * 64 + mt * 16 + r + 8 * hi;
        float val = acc[mt][nt][r] + bv;
        if (MODE == 2) {
          outF[(size_t)m * DM + n] = val;
        } else {
          int s = m >> 1, bb = m & 1;  // m = s*NB + b, NB==2
          int h = n >> 6, dh = n & 63;
          if (MODE == 0)
            outH[(((size_t)(h * NB + bb) * SEQ) + s) * DHD + dh] = (f16)val;
          else
            outH[(((size_t)(h * NB + bb) * DHD) + dh) * SEQ + s] = (f16)val;
        }
      }
    }
}

// ---------------------------------------------------------------------------
// Scores: S[q,j] = scale * sum_d Q[q,d]*Kt[d,j], masked; raw scores to attn buf
__global__ __launch_bounds__(256) void gemm_scores_kernel(const f16* __restrict__ Qh,
                                                          const f16* __restrict__ Kt,
                                                          const int* __restrict__ mask,
                                                          float* __restrict__ attn) {
  __shared__ __align__(16) f16 As[2][128 * 32];
  __shared__ __align__(16) f16 Bs[2][32 * 128];
  const int tid = threadIdx.x;
  const int lane = tid & 31, wave = tid >> 5;
  const int waveM = wave >> 2, waveN = wave & 3;
  const int hi = lane >> 4;
  const int m0 = blockIdx.y * 128, n0 = blockIdx.x * 128;
  const int hb = blockIdx.z;
  const f16* Q = Qh + (size_t)hb * SEQ * DHD;  // [S][64]
  const f16* K = Kt + (size_t)hb * DHD * SEQ;  // [64][S]

  auto issueA = [&](int kt, int buf) {
#pragma unroll
    for (int l = 0; l < 2; ++l) {
      int i = tid + l * 256;
      int r = i >> 2, c = (i & 3) * 8;
      async_b128(Q + (size_t)(m0 + r) * DHD + kt * 32 + c, &As[buf][r * 32 + c]);
    }
  };
  auto issueB = [&](int kt, int buf) {
#pragma unroll
    for (int l = 0; l < 2; ++l) {
      int i = tid + l * 256;
      int r = i >> 4, c = (i & 15) * 8;
      async_b128(K + (size_t)(kt * 32 + r) * SEQ + n0 + c, &Bs[buf][r * 128 + c]);
    }
  };

  const v8f vzero = {0, 0, 0, 0, 0, 0, 0, 0};
  v8f acc[4][2];
#pragma unroll
  for (int i = 0; i < 4; ++i)
#pragma unroll
    for (int j = 0; j < 2; ++j) acc[i][j] = vzero;

  issueA(0, 0);
  issueB(0, 0);
  const int NK = DHD / 32;  // 2
  for (int kt = 0; kt < NK; ++kt) {
    wait_async0();
    __syncthreads();
    if (kt + 1 < NK) {
      issueA(kt + 1, (kt + 1) & 1);
      issueB(kt + 1, (kt + 1) & 1);
    }
    const f16* Ac = As[kt & 1];
    const f16* Bc = Bs[kt & 1];

    Frag16 a[4], bf[2];
#pragma unroll
    for (int mt = 0; mt < 4; ++mt)
      load_a_frag(Ac, waveM * 64 + mt * 16 + (lane & 15), hi, a[mt]);
#pragma unroll
    for (int nt = 0; nt < 2; ++nt)
      load_b_frag<128>(Bc, lane, waveN * 32 + nt * 16, bf[nt]);
#pragma unroll
    for (int mt = 0; mt < 4; ++mt)
#pragma unroll
      for (int nt = 0; nt < 2; ++nt)
        acc[mt][nt] = wmma_f16f32(a[mt], bf[nt], acc[mt][nt]);
  }

  const float scale = 0.125f;  // 1/sqrt(64)
#pragma unroll
  for (int mt = 0; mt < 4; ++mt)
#pragma unroll
    for (int nt = 0; nt < 2; ++nt) {
      int j = n0 + waveN * 32 + nt * 16 + (lane & 15);
#pragma unroll
      for (int r = 0; r < 8; ++r) {
        int qi = m0 + waveM * 64 + mt * 16 + r + 8 * hi;
        float val = acc[mt][nt][r] * scale;
        val = mask[(size_t)qi * SEQ + j] ? val : -1e9f;
        attn[((size_t)hb * SEQ + qi) * SEQ + j] = val;
      }
    }
}

// ---------------------------------------------------------------------------
// Row softmax over 2048, values register-resident (1 read + 1 write / element)
__global__ __launch_bounds__(256) void softmax_kernel(float* __restrict__ attn) {
  __shared__ float red[256];
  const int tid = threadIdx.x;
  float* row = attn + ((size_t)blockIdx.y * SEQ + blockIdx.x) * SEQ;
  float v[8];
  float mx = -3.0e38f;
#pragma unroll
  for (int t = 0; t < 8; ++t) {
    v[t] = row[t * 256 + tid];
    mx = fmaxf(mx, v[t]);
  }
  red[tid] = mx;
  __syncthreads();
  for (int s = 128; s > 0; s >>= 1) {
    if (tid < s) red[tid] = fmaxf(red[tid], red[tid + s]);
    __syncthreads();
  }
  mx = red[0];
  __syncthreads();
  float sum = 0.f;
#pragma unroll
  for (int t = 0; t < 8; ++t) {
    v[t] = __expf(v[t] - mx);
    sum += v[t];
  }
  red[tid] = sum;
  __syncthreads();
  for (int s = 128; s > 0; s >>= 1) {
    if (tid < s) red[tid] += red[tid + s];
    __syncthreads();
  }
  float inv = 1.0f / red[0];
#pragma unroll
  for (int t = 0; t < 8; ++t) row[t * 256 + tid] = v[t] * inv;
}

// ---------------------------------------------------------------------------
// PV: O[q,dh] = sum_j P[q,j]*V[j,dh]; P fp32 -> f16 while staging to LDS;
// V tile staged with async b128. Block tile 128x64, BK=32, wave tile 32x32.
__global__ __launch_bounds__(256) void gemm_pv_kernel(const float* __restrict__ attn,
                                                      const f16* __restrict__ Vh,
                                                      f16* __restrict__ Ctx) {
  __shared__ __align__(16) f16 As[128 * 32];
  __shared__ __align__(16) f16 Bs[32 * 64];
  const int tid = threadIdx.x;
  const int lane = tid & 31, wave = tid >> 5;
  const int waveM = wave >> 1, waveN = wave & 1;
  const int hi = lane >> 4;
  const int m0 = blockIdx.x * 128;
  const int hb = blockIdx.y, h = hb >> 1, bb = hb & 1;
  const float* P = attn + (size_t)hb * SEQ * SEQ;
  const f16* V = Vh + (size_t)hb * SEQ * DHD;

  const v8f vzero = {0, 0, 0, 0, 0, 0, 0, 0};
  v8f acc[2][2];
#pragma unroll
  for (int i = 0; i < 2; ++i)
#pragma unroll
    for (int j = 0; j < 2; ++j) acc[i][j] = vzero;

  for (int kt = 0; kt < SEQ / 32; ++kt) {
    {  // V tile 32x64 halves: 256 x b128, async (overlaps with A staging below)
      int r = tid >> 3, c = (tid & 7) * 8;
      async_b128(V + (size_t)(kt * 32 + r) * DHD + c, &Bs[r * 64 + c]);
    }
#pragma unroll
    for (int l = 0; l < 4; ++l) {  // A: 128x32 fp32 -> f16
      int i = tid + l * 256;
      int r = i >> 3, c = (i & 7) * 4;
      float4 f = *(const float4*)(P + (size_t)(m0 + r) * SEQ + kt * 32 + c);
      union { f16 h[4]; uint2 u; } pk;
      pk.h[0] = (f16)f.x; pk.h[1] = (f16)f.y; pk.h[2] = (f16)f.z; pk.h[3] = (f16)f.w;
      *(uint2*)(As + r * 32 + c) = pk.u;
    }
    wait_async0();
    __syncthreads();

    Frag16 a[2], bf[2];
#pragma unroll
    for (int mt = 0; mt < 2; ++mt)
      load_a_frag(As, waveM * 32 + mt * 16 + (lane & 15), hi, a[mt]);
#pragma unroll
    for (int nt = 0; nt < 2; ++nt)
      load_b_frag<64>(Bs, lane, waveN * 32 + nt * 16, bf[nt]);
#pragma unroll
    for (int mt = 0; mt < 2; ++mt)
#pragma unroll
      for (int nt = 0; nt < 2; ++nt)
        acc[mt][nt] = wmma_f16f32(a[mt], bf[nt], acc[mt][nt]);
    __syncthreads();
  }

#pragma unroll
  for (int mt = 0; mt < 2; ++mt)
#pragma unroll
    for (int nt = 0; nt < 2; ++nt) {
      int dh = waveN * 32 + nt * 16 + (lane & 15);
#pragma unroll
      for (int r = 0; r < 8; ++r) {
        int qi = m0 + waveM * 32 + mt * 16 + r + 8 * hi;
        Ctx[((size_t)qi * NB + bb) * DM + h * DHD + dh] = (f16)acc[mt][nt][r];
      }
    }
}

// ---------------------------------------------------------------------------
extern "C" void kernel_launch(void* const* d_in, const int* in_sizes, int n_in,
                              void* d_out, int out_size, void* d_ws, size_t ws_size,
                              hipStream_t stream) {
  (void)in_sizes; (void)n_in; (void)out_size; (void)ws_size;
  const float* q    = (const float*)d_in[0];
  const float* k    = (const float*)d_in[1];
  const float* v    = (const float*)d_in[2];
  const int*   mask = (const int*)d_in[3];
  const float* Wq = (const float*)d_in[4];  const float* bq = (const float*)d_in[5];
  const float* Wk = (const float*)d_in[6];  const float* bk = (const float*)d_in[7];
  const float* Wv = (const float*)d_in[8];  const float* bv = (const float*)d_in[9];
  const float* Wo = (const float*)d_in[10]; const float* bo = (const float*)d_in[11];

  const size_t NX = (size_t)MROWS * DM;  // 4M halves per activation tensor
  const size_t NW = (size_t)DM * DM;     // 1M halves per weight
  f16* Xq  = (f16*)d_ws;
  f16* Xk  = Xq + NX;
  f16* Xv  = Xk + NX;
  f16* WqT = Xv + NX;
  f16* WkT = WqT + NW;
  f16* WvT = WkT + NW;
  f16* WoT = WvT + NW;
  f16* Qh  = WoT + NW;       // [H][B][S][64]
  f16* Kt  = Qh + NX;        // [H][B][64][S]
  f16* Vh  = Kt + NX;        // [H][B][S][64]
  f16* Ctx = Vh + NX;        // [M][D]
  float* outF = (float*)d_out;
  float* attn = outF + NX;   // [H][B][S][S]

  // 1) pack activations fp32 -> f16 (4/thread)
  cvt_f32_f16x4_kernel<<<(int)(NX / 1024), 256, 0, stream>>>((const float4*)q, (uint2*)Xq);
  cvt_f32_f16x4_kernel<<<(int)(NX / 1024), 256, 0, stream>>>((const float4*)k, (uint2*)Xk);
  cvt_f32_f16x4_kernel<<<(int)(NX / 1024), 256, 0, stream>>>((const float4*)v, (uint2*)Xv);

  // 2) transpose-pack weights so GEMM B tiles are k-major
  transpose_pack_kernel<<<dim3(32, 32), 256, 0, stream>>>(Wq, WqT);
  transpose_pack_kernel<<<dim3(32, 32), 256, 0, stream>>>(Wk, WkT);
  transpose_pack_kernel<<<dim3(32, 32), 256, 0, stream>>>(Wv, WvT);
  transpose_pack_kernel<<<dim3(32, 32), 256, 0, stream>>>(Wo, WoT);

  // 3) Q/K/V projections (K written transposed per (h,b))
  gemm_proj_kernel<0><<<dim3(8, 32), 256, 0, stream>>>(Xq, WqT, bq, Qh, nullptr);
  gemm_proj_kernel<1><<<dim3(8, 32), 256, 0, stream>>>(Xk, WkT, bk, Kt, nullptr);
  gemm_proj_kernel<0><<<dim3(8, 32), 256, 0, stream>>>(Xv, WvT, bv, Vh, nullptr);

  // 4) scores = scale * Q K^T (masked), raw into attn region of d_out
  gemm_scores_kernel<<<dim3(16, 16, 32), 256, 0, stream>>>(Qh, Kt, mask, attn);

  // 5) softmax in place (attn region of d_out == attn_weights output)
  softmax_kernel<<<dim3(SEQ, NH * NB), 256, 0, stream>>>(attn);

  // 6) context = P @ V
  gemm_pv_kernel<<<dim3(16, 32), 256, 0, stream>>>(attn, Vh, Ctx);

  // 7) output projection -> d_out fp32 [S,B,D]
  gemm_proj_kernel<2><<<dim3(8, 32), 256, 0, stream>>>(Ctx, WoT, bo, nullptr, outF);
}
